// SoftCapAttention_3066606650060
// MI455X (gfx1250) — matrix operand order repaired and checked by reading, not compiled, not run on hardware
//
#include <hip/hip_runtime.h>
#include <hip/hip_bf16.h>

typedef __attribute__((ext_vector_type(16))) _Float16 v16h;
typedef __attribute__((ext_vector_type(8)))  float    v8f;
typedef __attribute__((ext_vector_type(4)))  float    v4f;

#define N_CTX   2048
#define DHEAD   128
#define BLOCK_M 128          // query rows per workgroup (8 waves x 16)
#define BLOCK_N 32           // KV rows per step (matches WMMA K=32)
#define KPAD    144          // 128 + pad halves; row = 288 B (32B aligned)
#define VPAD    48           // 32 + pad halves;  row = 96 B  (32B aligned)
#define SOFTCAPV 20.0f

__global__ __launch_bounds__(256, 1)
void softcap_flash_attn_gfx1250(const float* __restrict__ Qg,
                                const float* __restrict__ Kg,
                                const float* __restrict__ Vg,
                                float* __restrict__ Og)
{
    // double-buffered K/V tiles so next tile's HBM loads overlap compute
    __shared__ __align__(32) _Float16 Kc[2][BLOCK_N][KPAD];  // K tile, row-major f16
    __shared__ __align__(32) _Float16 Vt[2][DHEAD][VPAD];    // V tile, transposed f16
    __shared__ __align__(32) _Float16 Pb[8][16][36];         // per-wave P transpose scratch

    const int tid  = threadIdx.x;
    const int lane = tid & 31;
    const int wave = tid >> 5;
    const int half = lane >> 4;     // 0 = lanes 0-15, 1 = lanes 16-31
    const int ln16 = lane & 15;

    const int blocks_per_head = N_CTX / BLOCK_M;          // 16
    const int bh = blockIdx.x / blocks_per_head;          // fused B*H index
    const int qb = blockIdx.x % blocks_per_head;

    const size_t head_off = (size_t)bh * N_CTX * DHEAD;
    const float* Qh = Qg + head_off;
    const float* Kh = Kg + head_off;
    const float* Vh = Vg + head_off;
    float*       Oh = Og + head_off;

    // staging geometry: 256 threads x 16 floats = one 32x128 tile
    const int srow = (tid * 16) >> 7;
    const int scol = (tid * 16) & 127;

    v4f kr[4], vr[4];   // in-flight staging registers (software pipeline)

    auto issue_loads = [&](int j) {
        const float* kp = Kh + (size_t)(j + srow) * DHEAD + scol;
        const float* vp = Vh + (size_t)(j + srow) * DHEAD + scol;
#pragma unroll
        for (int i = 0; i < 4; ++i) {
            kr[i] = *(const v4f*)(kp + 4 * i);
            vr[i] = *(const v4f*)(vp + 4 * i);
        }
    };
    auto store_tile = [&](int b) {
#pragma unroll
        for (int i = 0; i < 4; ++i) {
#pragma unroll
            for (int c = 0; c < 4; ++c) {
                Kc[b][srow][scol + 4 * i + c] = (_Float16)kr[i][c];
                Vt[b][scol + 4 * i + c][srow] = (_Float16)vr[i][c];   // transpose
            }
        }
    };

    // ---- load this wave's 16x128 Q tile as four 16x32 f16 A-fragments ----
    // A operand as packed here (contiguous k, not AMD's table order): lanes 0-15 row M=lane, K = 0..15;
    //                           lanes 16-31 row M=lane-16, K = 16..31.
    const int qrow = qb * BLOCK_M + wave * 16 + ln16;
    v16h qa[4];
#pragma unroll
    for (int f = 0; f < 4; ++f) {
        const float* qp = Qh + (size_t)qrow * DHEAD + f * 32;
#pragma unroll
        for (int e = 0; e < 16; ++e) {
            int k = e + 16 * half;
            qa[f][e] = (_Float16)qp[k];
        }
    }

    v8f o[8];
#pragma unroll
    for (int c = 0; c < 8; ++c) o[c] = {};
    float m_i[8], l_i[8];
#pragma unroll
    for (int r = 0; r < 8; ++r) { m_i[r] = -1e30f; l_i[r] = 0.0f; }

    const float scale  = 0.08838834764831845f;  // 1/sqrt(128)
    const float invcap = 1.0f / SOFTCAPV;

    // prologue: fill buffer 0
    issue_loads(0);
    store_tile(0);
    __syncthreads();

    for (int j = 0, it = 0; j < N_CTX; j += BLOCK_N, ++it) {
        const int  cur  = it & 1;
        const bool more = (j + BLOCK_N) < N_CTX;

        // issue next tile's global loads now; they complete during compute
        if (more) {
            issue_loads(j + BLOCK_N);
            if (j + 2 * BLOCK_N < N_CTX) {              // global_prefetch_b8
                __builtin_prefetch(Kh + (size_t)(j + 2 * BLOCK_N + srow) * DHEAD + scol, 0, 1);
                __builtin_prefetch(Vh + (size_t)(j + 2 * BLOCK_N + srow) * DHEAD + scol, 0, 1);
            }
        }

        // ---- S = Q K^T : batch all 8 B-fragments, then 8 WMMAs ----
        // B operand as packed here (contiguous k, same order as A): lane holds column n=lane%16, 16 contiguous K.
        v16h bk[8];
#pragma unroll
        for (int f = 0; f < 4; ++f) {
            bk[2 * f]     = *(const v16h*)&Kc[cur][ln16     ][32 * f + 16 * half];
            bk[2 * f + 1] = *(const v16h*)&Kc[cur][16 + ln16][32 * f + 16 * half];
        }
        v8f s0 = {}, s1 = {};
#pragma unroll
        for (int f = 0; f < 4; ++f) {
            s0 = __builtin_amdgcn_wmma_f32_16x16x32_f16(false, qa[f], false, bk[2 * f],
                                                        (short)0, s0, false, false);
            s1 = __builtin_amdgcn_wmma_f32_16x16x32_f16(false, qa[f], false, bk[2 * f + 1],
                                                        (short)0, s1, false, false);
        }

        // ---- tanh soft-cap + online softmax (rows live in 16-lane groups) ----
#pragma unroll
        for (int r = 0; r < 8; ++r) {
            float x0 = s0[r] * scale;
            float x1 = s1[r] * scale;
            float t0 = __expf(2.0f * x0 * invcap);
            float t1 = __expf(2.0f * x1 * invcap);
            x0 = SOFTCAPV * ((t0 - 1.0f) / (t0 + 1.0f));
            x1 = SOFTCAPV * ((t1 - 1.0f) / (t1 + 1.0f));

            float mx = fmaxf(x0, x1);
            mx = fmaxf(mx, __shfl_xor(mx, 1, 16));
            mx = fmaxf(mx, __shfl_xor(mx, 2, 16));
            mx = fmaxf(mx, __shfl_xor(mx, 4, 16));
            mx = fmaxf(mx, __shfl_xor(mx, 8, 16));
            float mnew = fmaxf(m_i[r], mx);

            float e0 = __expf(x0 - mnew);
            float e1 = __expf(x1 - mnew);
            float rs = e0 + e1;
            rs += __shfl_xor(rs, 1, 16);
            rs += __shfl_xor(rs, 2, 16);
            rs += __shfl_xor(rs, 4, 16);
            rs += __shfl_xor(rs, 8, 16);

            float sf = __expf(m_i[r] - mnew);
            l_i[r] = l_i[r] * sf + rs;
            m_i[r] = mnew;
#pragma unroll
            for (int c = 0; c < 8; ++c) o[c][r] = o[c][r] * sf;

            int prow = r + 8 * half;                   // C-layout row of this VGPR
            Pb[wave][prow][ln16]      = (_Float16)e0;
            Pb[wave][prow][16 + ln16] = (_Float16)e1;
        }

        // wave-local C-layout -> A-layout transpose through LDS
        __builtin_amdgcn_wave_barrier();
        asm volatile("s_wait_dscnt 0" ::: "memory");
        __builtin_amdgcn_wave_barrier();

        v16h pa;
#pragma unroll
        for (int e = 0; e < 16; ++e) {
            int k = e + 16 * half;
            pa[e] = Pb[wave][ln16][k];
        }

        // ---- O += P V : batch 8 V-fragments, then 8 WMMAs ----
        v16h bv[8];
#pragma unroll
        for (int c = 0; c < 8; ++c)
            bv[c] = *(const v16h*)&Vt[cur][c * 16 + ln16][16 * half];
#pragma unroll
        for (int c = 0; c < 8; ++c)
            o[c] = __builtin_amdgcn_wmma_f32_16x16x32_f16(false, pa, false, bv[c],
                                                          (short)0, o[c], false, false);

        // commit next tile into the alternate buffer
        if (more) {
            __syncthreads();           // all waves done reading buffer (cur^1)
            store_tile(cur ^ 1);
            __syncthreads();           // next tile visible to all waves
        }
    }

    // ---- normalize and store ----
#pragma unroll
    for (int r = 0; r < 8; ++r) {
        float inv = 1.0f / l_i[r];
        int row = qb * BLOCK_M + wave * 16 + r + 8 * half;
        float* op = Oh + (size_t)row * DHEAD + ln16;
#pragma unroll
        for (int c = 0; c < 8; ++c) op[c * 16] = o[c][r] * inv;
    }
}

extern "C" void kernel_launch(void* const* d_in, const int* in_sizes, int n_in,
                              void* d_out, int out_size, void* d_ws, size_t ws_size,
                              hipStream_t stream) {
    (void)in_sizes; (void)n_in; (void)out_size; (void)d_ws; (void)ws_size;
    const float* Q = (const float*)d_in[0];
    const float* K = (const float*)d_in[1];
    const float* V = (const float*)d_in[2];
    float*       O = (float*)d_out;

    const int heads = 32;                        // B*H = 2*16
    const int blocks_per_head = N_CTX / BLOCK_M; // 16
    dim3 grid(heads * blocks_per_head);          // 512
    dim3 block(256);                             // 8 wave32
    softcap_flash_attn_gfx1250<<<grid, block, 0, stream>>>(Q, K, V, O);
}
